// LinearAttention_55619826483330
// MI455X (gfx1250) — matrix operand order repaired and checked by reading, not compiled, not run on hardware
//
#include <hip/hip_runtime.h>
#include <stdint.h>

// ---------------------------------------------------------------------------
// LinearAttention fused pipeline for gfx1250 (MI455X).
// bf16 WMMA GEMMs, CDNA5 async global->LDS copies, double-buffered LDS tiles
// so the next tile's copy overlaps the current tile's WMMAs (ASYNCcnt path).
// B=16, C=512, H=W=64 -> NPIX=65536 pixels, heads=8, dim_head=64.
// ---------------------------------------------------------------------------

#define NPIX     65536L   // total pixels (B * H * W)
#define NPP      4096     // pixels per batch image (H*W)
#define CIN      512
#define CQKV     1536
#define QK_SCALE 0.125f   // 64^-0.5
#define LN_EPS   1e-5f

typedef __attribute__((ext_vector_type(16))) __bf16 bf16x16;
typedef __attribute__((ext_vector_type(8)))  __bf16 bf16x8;
typedef __attribute__((ext_vector_type(8)))  float  f32x8;
typedef __attribute__((ext_vector_type(4)))  float  f32x4;

// ---------------- CDNA5 async global->LDS copy (ASYNCcnt path) --------------
// 16 bytes per lane; LDS destination is a byte offset from the wave LDS base
// (each kernel below uses exactly ONE __shared__ pool, allocated at offset 0).
__device__ __forceinline__ void async_copy_b128(uint32_t lds_off, const void* g) {
    asm volatile("global_load_async_to_lds_b128 %0, %1, off"
                 :: "v"(lds_off), "v"((uint64_t)(uintptr_t)g)
                 : "memory");
}
__device__ __forceinline__ void async_wait0() {
    asm volatile("s_wait_asynccnt 0" ::: "memory");
}

// ---------------- WMMA fragment helpers (wave32, 16x16x32 bf16) -------------
// A fragment: 16(M) x 32(K) tile, row-major in LDS with leading dim `ld`
// (elements). Lane l<16 holds row l, K = kb..kb+7 and kb+16..kb+23 (kb=0);
// lanes 16-31 hold kb=8.  Two aligned 16B loads per lane.
__device__ __forceinline__ bf16x16 frag_ld_rowmajor(const __bf16* t, int ld) {
    int lane = threadIdx.x & 31;
    int r  = lane & 15;
    int kb = (lane >> 4) << 3;                // 0 or 8
    const __bf16* p = t + r * ld + kb;
    union { bf16x16 v; bf16x8 h[2]; } u;
    u.h[0] = *(const bf16x8*)(p);
    u.h[1] = *(const bf16x8*)(p + 16);
    return u.v;
}

// B fragment: 32(K) x 16(N) tile stored COLUMN-major in LDS (each column's K
// run contiguous, leading dim `ld`). Lane l holds column (l&15), K = kb..kb+7
// and kb+16..kb+23 with kb = (l>>4)*8.  Two aligned 16B loads.
__device__ __forceinline__ bf16x16 frag_ld_colmajor(const __bf16* t, int ld) {
    int lane = threadIdx.x & 31;
    int c  = lane & 15;
    int kc = (lane >> 4) << 3;                // 0 or 8
    const __bf16* p = t + c * ld + kc;
    union { bf16x16 v; bf16x8 h[2]; } u;
    u.h[0] = *(const bf16x8*)(p);
    u.h[1] = *(const bf16x8*)(p + 16);
    return u.v;
}

__device__ __forceinline__ f32x8 wmma_bf16(bf16x16 a, bf16x16 b, f32x8 c) {
    return __builtin_amdgcn_wmma_f32_16x16x32_bf16(false, a, false, b,
                                                   (short)0, c, false, false);
}

// Shared 128x128 compute body: 2x4 fragments per wave from A/B LDS tiles.
__device__ __forceinline__ void gemm_compute(const __bf16* lA, const __bf16* lB,
                                             int wm, int wn, f32x8 acc[2][4]) {
    bf16x16 af0 = frag_ld_rowmajor(lA + (wm * 32 + 0)  * 40, 40);
    bf16x16 af1 = frag_ld_rowmajor(lA + (wm * 32 + 16) * 40, 40);
    #pragma unroll
    for (int j = 0; j < 4; ++j) {
        bf16x16 bfv = frag_ld_colmajor(lB + (wn * 64 + j * 16) * 40, 40);
        acc[0][j] = wmma_bf16(af0, bfv, acc[0][j]);
        acc[1][j] = wmma_bf16(af1, bfv, acc[1][j]);
    }
}

// ---------------- small utility kernels -------------------------------------
__global__ void __launch_bounds__(256) cvt_bf16_kernel(const float* __restrict__ a,
                                                       __bf16* __restrict__ o, int n) {
    int i = blockIdx.x * 256 + threadIdx.x;
    if (i < n) o[i] = (__bf16)a[i];
}

// LN stats over C for x layout [b, c, n]; one thread per pixel p = b*4096+n.
__global__ void __launch_bounds__(256) ln_stats_x_kernel(const float* __restrict__ x,
                                                         float* __restrict__ mean,
                                                         float* __restrict__ rstd) {
    long p = (long)blockIdx.x * 256 + threadIdx.x;
    long b = p >> 12, n = p & 4095;
    const float* base = x + (b << 21) + n;    // b*512*4096 + n
    float s = 0.f, s2 = 0.f;
    #pragma unroll 8
    for (int c = 0; c < CIN; ++c) {
        float v = base[(long)c << 12];
        s += v; s2 += v * v;
    }
    float m   = s * (1.f / CIN);
    float var = s2 * (1.f / CIN) - m * m;
    mean[p] = m;
    rstd[p] = rsqrtf(var + LN_EPS);
}

// LN stats over C for pre layout [c, p] (p contiguous).
__global__ void __launch_bounds__(256) ln_stats_pre_kernel(const float* __restrict__ pre,
                                                           float* __restrict__ mean,
                                                           float* __restrict__ rstd) {
    long p = (long)blockIdx.x * 256 + threadIdx.x;
    const float* base = pre + p;
    float s = 0.f, s2 = 0.f;
    #pragma unroll 8
    for (int c = 0; c < CIN; ++c) {
        float v = base[(long)c << 16];
        s += v; s2 += v * v;
    }
    float m   = s * (1.f / CIN);
    float var = s2 * (1.f / CIN) - m * m;
    mean[p] = m;
    rstd[p] = rsqrtf(var + LN_EPS);
}

// ---------------- GEMM1: qkv = w_qkv (bf16) x LN(x) -------------------------
// Output layout: qkv_bf[o * 65536 + p], o in [0,1536), p = b*4096 + n.
// A tile (weights) via async global->LDS; LN normalize fused into the B-tile
// load (x stays f32 and lives in the 192MB L2).  Double-buffered LDS tiles.
__device__ __forceinline__ void qkv_load_tile(
        const __bf16* wq, const float* x, const float* g_ln,
        const float* sMean, const float* sRstd, __bf16* lB, uint32_t ldsOffA,
        int m0, long b, long n0, int k0, int tid) {
    // A tile: async copy, 2x16B per thread
    int row = tid >> 1, kh = (tid & 1) << 4;
    const __bf16* ga = wq + (long)(m0 + row) * CIN + k0 + kh;
    uint32_t lo = ldsOffA + (uint32_t)(row * 40 + kh) * 2u;
    async_copy_b128(lo,      ga);
    async_copy_b128(lo + 16, ga + 8);
    // B tile: x[c][128 px], normalized, scattered col-major (overlaps async)
    int   cl = tid >> 3;
    int   px = (tid & 7) << 4;
    int   c  = k0 + cl;
    float gl = g_ln[c];
    const float* xg = x + (b << 21) + ((long)c << 12) + n0 + px;
    __builtin_prefetch(xg + (32L << 12), 0, 3);   // speculative: next k-step
    #pragma unroll
    for (int i = 0; i < 16; i += 4) {
        f32x4 v = *(const f32x4*)(xg + i);
        #pragma unroll
        for (int j = 0; j < 4; ++j) {
            int pp = px + i + j;
            lB[pp * 40 + cl] = (__bf16)((v[j] - sMean[pp]) * sRstd[pp] * gl);
        }
    }
}

__global__ void __launch_bounds__(256) gemm_qkv_kernel(
        const float* __restrict__ x, const float* __restrict__ mean,
        const float* __restrict__ rstd, const float* __restrict__ g_ln,
        const __bf16* __restrict__ wq, __bf16* __restrict__ out) {
    // one LDS pool so async byte offsets are well-defined (pool base = 0):
    //   A0 @ 0, B0 @ 10240, A1 @ 20480, B1 @ 30720, stats @ 40960
    __shared__ __align__(16) char smem[41984];
    __bf16* lA0   = (__bf16*)smem;
    __bf16* lB0   = (__bf16*)(smem + 10240);
    __bf16* lA1   = (__bf16*)(smem + 20480);
    __bf16* lB1   = (__bf16*)(smem + 30720);
    float*  sMean = (float*)(smem + 40960);
    float*  sRstd = (float*)(smem + 41472);

    int  tid  = threadIdx.x;
    long p0   = (long)blockIdx.x * 128;
    int  m0   = blockIdx.y * 128;
    long b    = p0 >> 12;
    long n0   = p0 & 4095;

    if (tid < 128) { sMean[tid] = mean[p0 + tid]; sRstd[tid] = rstd[p0 + tid]; }
    __syncthreads();

    int w = tid >> 5, wm = w & 3, wn = w >> 2;
    f32x8 acc[2][4] = {};

    // prologue: tile 0 into buffer 0
    qkv_load_tile(wq, x, g_ln, sMean, sRstd, lB0, 0u, m0, b, n0, 0, tid);
    async_wait0();
    __syncthreads();

    for (int k0 = 0; k0 < CIN; k0 += 64) {
        if (k0 + 32 < CIN)
            qkv_load_tile(wq, x, g_ln, sMean, sRstd, lB1, 20480u,
                          m0, b, n0, k0 + 32, tid);
        gemm_compute(lA0, lB0, wm, wn, acc);
        async_wait0();
        __syncthreads();

        if (k0 + 64 < CIN)
            qkv_load_tile(wq, x, g_ln, sMean, sRstd, lB0, 0u,
                          m0, b, n0, k0 + 64, tid);
        gemm_compute(lA1, lB1, wm, wn, acc);
        async_wait0();
        __syncthreads();
    }

    // Epilogue: D layout -> lane l<16: (M=r, N=l); l>=16: (M=r+8, N=l-16).
    int lane = tid & 31;
    int rowOff = (lane >> 4) << 3, colOff = lane & 15;
    #pragma unroll
    for (int i = 0; i < 2; ++i)
        #pragma unroll
        for (int j = 0; j < 4; ++j) {
            long orow = m0 + wm * 32 + i * 16 + rowOff;
            long col  = p0 + wn * 64 + j * 16 + colOff;
            #pragma unroll
            for (int r = 0; r < 8; ++r)
                out[(orow + r) * NPIX + col] = (__bf16)acc[i][j][r];
        }
}

// ---------------- softmax over d for q (in place, bf16) ---------------------
__global__ void __launch_bounds__(256) softmax_q_kernel(__bf16* __restrict__ qkv) {
    long t = (long)blockIdx.x * 256 + threadIdx.x;   // [0, 8*65536)
    long h = t >> 16, p = t & 65535;
    __bf16* base = qkv + (h << 6) * NPIX + p;
    float mx = -1e30f;
    #pragma unroll 8
    for (int d = 0; d < 64; ++d) mx = fmaxf(mx, (float)base[(long)d * NPIX]);
    float s = 0.f;
    #pragma unroll 8
    for (int d = 0; d < 64; ++d) s += __expf((float)base[(long)d * NPIX] - mx);
    float inv = QK_SCALE / s;
    #pragma unroll 8
    for (int d = 0; d < 64; ++d) {
        long idx = (long)d * NPIX;
        base[idx] = (__bf16)(__expf((float)base[idx] - mx) * inv);
    }
}

// ---------------- softmax over n for k (in place, bf16) ---------------------
__global__ void __launch_bounds__(256) softmax_k_kernel(__bf16* __restrict__ qkv) {
    int bx = blockIdx.x;                  // [0, 16*512)
    int b  = bx >> 9, r = bx & 511;
    __bf16* base = qkv + (long)(CIN + r) * NPIX + ((long)b << 12);
    __shared__ float red[256];
    int tid = threadIdx.x;

    float mx = -1e30f;
    for (int i = tid; i < NPP; i += 256) mx = fmaxf(mx, (float)base[i]);
    red[tid] = mx; __syncthreads();
    #pragma unroll
    for (int s = 128; s > 0; s >>= 1) {
        if (tid < s) red[tid] = fmaxf(red[tid], red[tid + s]);
        __syncthreads();
    }
    mx = red[0]; __syncthreads();

    float sum = 0.f;
    for (int i = tid; i < NPP; i += 256) sum += __expf((float)base[i] - mx);
    red[tid] = sum; __syncthreads();
    #pragma unroll
    for (int s = 128; s > 0; s >>= 1) {
        if (tid < s) red[tid] += red[tid + s];
        __syncthreads();
    }
    float inv = 1.f / red[0];
    for (int i = tid; i < NPP; i += 256)
        base[i] = (__bf16)(__expf((float)base[i] - mx) * inv);
}

// ---------------- fused attention per (b, h) --------------------------------
// Phase 1: context[d,e] = (1/n) * sum_n k[d,n] v[e,n]  (64x64, K=4096)
// Phase 2: out[e,n]    = sum_d context[d,e] q[d,n]      (64x4096, K=64)
// Phase-1 k/v tiles are pure copies -> async global->LDS, double-buffered.
__device__ __forceinline__ void attn_load_kv(const __bf16* qkv, long kBase,
                                             long vBase, int n0, uint32_t off,
                                             int tid) {
    int row = tid >> 2, ch = (tid & 3) << 3;
    uint32_t lo = off + (uint32_t)(row * 40 + ch) * 2u;
    async_copy_b128(lo,         qkv + kBase + (long)row * NPIX + n0 + ch);
    async_copy_b128(lo + 5120u, qkv + vBase + (long)row * NPIX + n0 + ch);
}

__device__ __forceinline__ void attn_p1_compute(const __bf16* lA, const __bf16* lB,
                                                int wm, int wn2, f32x8 acc[2]) {
    bf16x16 af = frag_ld_rowmajor(lA + (wm * 16) * 40, 40);
    #pragma unroll
    for (int j = 0; j < 2; ++j) {
        bf16x16 bfv = frag_ld_colmajor(lB + (wn2 * 32 + j * 16) * 40, 40);
        acc[j] = wmma_bf16(af, bfv, acc[j]);
    }
}

__global__ void __launch_bounds__(256) attention_kernel(
        const __bf16* __restrict__ qkv, __bf16* __restrict__ attn) {
    // pool: K0 @ 0, V0 @ 5120, K1 @ 10240, V1 @ 15360, sCtx @ 20480
    // phase 2: lQ (128x72 bf16 = 18432B) reuses the ping-pong region @ 0
    __shared__ __align__(16) char smem[29696];
    __bf16* lA0  = (__bf16*)smem;
    __bf16* lB0  = (__bf16*)(smem + 5120);
    __bf16* lA1  = (__bf16*)(smem + 10240);
    __bf16* lB1  = (__bf16*)(smem + 15360);
    __bf16* lQ   = (__bf16*)smem;
    __bf16* sCtx = (__bf16*)(smem + 20480);   // 64 x 72 context^T, bf16

    int b = blockIdx.x >> 3, h = blockIdx.x & 7;
    long kBase = ((long)(CIN      + h * 64)) * NPIX + ((long)b << 12);
    long vBase = ((long)(2 * CIN  + h * 64)) * NPIX + ((long)b << 12);
    long qBase = ((long)(            h * 64)) * NPIX + ((long)b << 12);

    int tid = threadIdx.x, w = tid >> 5, lane = tid & 31;
    int wm = w & 3, wn2 = w >> 2;            // phase1: 4x2 wave grid
    int rowOff = (lane >> 4) << 3, colOff = lane & 15;

    // ---- phase 1 (double-buffered over n) ----
    f32x8 acc[2] = {};
    attn_load_kv(qkv, kBase, vBase, 0, 0u, tid);
    async_wait0();
    __syncthreads();
    for (int n0 = 0; n0 < NPP; n0 += 64) {
        if (n0 + 32 < NPP) attn_load_kv(qkv, kBase, vBase, n0 + 32, 10240u, tid);
        attn_p1_compute(lA0, lB0, wm, wn2, acc);
        async_wait0();
        __syncthreads();

        if (n0 + 64 < NPP) attn_load_kv(qkv, kBase, vBase, n0 + 64, 0u, tid);
        attn_p1_compute(lA1, lB1, wm, wn2, acc);
        async_wait0();
        __syncthreads();
    }
    // store context TRANSPOSED (e-major) as bf16 into sCtx, scaled by 1/n
    #pragma unroll
    for (int j = 0; j < 2; ++j)
        #pragma unroll
        for (int r = 0; r < 8; ++r) {
            int d = wm * 16 + rowOff + r;
            int e = wn2 * 32 + j * 16 + colOff;
            sCtx[e * 72 + d] = (__bf16)(acc[j][r] * (1.f / NPP));
        }
    __syncthreads();

    // ---- phase 2 ----
    int wn = w >> 2;                         // 0..1
    for (int t0 = 0; t0 < NPP; t0 += 128) {
        // load q tile (d x 128 pixels) transposed into lQ
        int d = tid >> 2, px = (tid & 3) << 5;
        const __bf16* g = qkv + qBase + (long)d * NPIX + t0 + px;
        #pragma unroll
        for (int i = 0; i < 32; i += 16) {
            bf16x16 v = *(const bf16x16*)(g + i);
            #pragma unroll
            for (int j2 = 0; j2 < 16; ++j2)
                lQ[(px + i + j2) * 72 + d] = v[j2];
        }
        __syncthreads();
        f32x8 oacc[4] = {};
        #pragma unroll
        for (int k0 = 0; k0 < 64; k0 += 32) {
            bf16x16 af = frag_ld_rowmajor(sCtx + (wm * 16) * 72 + k0, 72);
            #pragma unroll
            for (int j = 0; j < 4; ++j) {
                bf16x16 bfv = frag_ld_colmajor(lQ + (wn * 64 + j * 16) * 72 + k0, 72);
                oacc[j] = wmma_bf16(af, bfv, oacc[j]);
            }
        }
        #pragma unroll
        for (int j = 0; j < 4; ++j)
            #pragma unroll
            for (int r = 0; r < 8; ++r) {
                int  e   = wm * 16 + rowOff + r;
                long col = t0 + wn * 64 + j * 16 + colOff;
                attn[((long)(h * 64 + e)) * NPIX + ((long)b << 12) + col]
                    = (__bf16)oacc[j][r];
            }
        __syncthreads();
    }
}

// ---------------- GEMM2: pre = w_out (bf16) x attn + b_out ------------------
__device__ __forceinline__ void out_load_tile(
        const __bf16* attnp, const __bf16* wo, __bf16* lB, uint32_t ldsOffA,
        int m0, long p0, int k0, int tid) {
    int row = tid >> 1, kh = (tid & 1) << 4;
    const __bf16* ga = wo + (long)(m0 + row) * CIN + k0 + kh;
    uint32_t lo = ldsOffA + (uint32_t)(row * 40 + kh) * 2u;
    async_copy_b128(lo,      ga);
    async_copy_b128(lo + 16, ga + 8);

    int cl = tid >> 3;
    int px = (tid & 7) << 4;
    const __bf16* g = attnp + (long)(k0 + cl) * NPIX + p0 + px;
    __builtin_prefetch(g + 32L * NPIX, 0, 3);     // speculative: next k-step
    bf16x16 v = *(const bf16x16*)(g);
    #pragma unroll
    for (int j = 0; j < 16; ++j)
        lB[(px + j) * 40 + cl] = v[j];
}

__global__ void __launch_bounds__(256) gemm_out_kernel(
        const __bf16* __restrict__ attn, const __bf16* __restrict__ wo,
        const float* __restrict__ bias, float* __restrict__ pre) {
    // A0 @ 0, B0 @ 10240, A1 @ 20480, B1 @ 30720
    __shared__ __align__(16) char smem[40960];
    __bf16* lA0 = (__bf16*)smem;
    __bf16* lB0 = (__bf16*)(smem + 10240);
    __bf16* lA1 = (__bf16*)(smem + 20480);
    __bf16* lB1 = (__bf16*)(smem + 30720);

    int  tid = threadIdx.x;
    long p0  = (long)blockIdx.x * 128;
    int  m0  = blockIdx.y * 128;

    int w = tid >> 5, wm = w & 3, wn = w >> 2;
    f32x8 acc[2][4] = {};

    out_load_tile(attn, wo, lB0, 0u, m0, p0, 0, tid);
    async_wait0();
    __syncthreads();

    for (int k0 = 0; k0 < CIN; k0 += 64) {
        if (k0 + 32 < CIN)
            out_load_tile(attn, wo, lB1, 20480u, m0, p0, k0 + 32, tid);
        gemm_compute(lA0, lB0, wm, wn, acc);
        async_wait0();
        __syncthreads();

        if (k0 + 64 < CIN)
            out_load_tile(attn, wo, lB0, 0u, m0, p0, k0 + 64, tid);
        gemm_compute(lA1, lB1, wm, wn, acc);
        async_wait0();
        __syncthreads();
    }

    int lane = tid & 31;
    int rowOff = (lane >> 4) << 3, colOff = lane & 15;
    #pragma unroll
    for (int i = 0; i < 2; ++i)
        #pragma unroll
        for (int j = 0; j < 4; ++j) {
            long orow = m0 + wm * 32 + i * 16 + rowOff;
            long col  = p0 + wn * 64 + j * 16 + colOff;
            #pragma unroll
            for (int r = 0; r < 8; ++r)
                pre[(orow + r) * NPIX + col] = acc[i][j][r] + bias[orow + r];
        }
}

// ---------------- final LN + residual ---------------------------------------
__global__ void __launch_bounds__(256) ln_residual_kernel(
        const float* __restrict__ pre, const float* __restrict__ mean,
        const float* __restrict__ rstd, const float* __restrict__ g_out,
        const float* __restrict__ x, float* __restrict__ out) {
    long i = (long)blockIdx.x * 256 + threadIdx.x;   // [0, 16*512*4096)
    long n = i & 4095;
    long c = (i >> 12) & 511;
    long b = i >> 21;
    long p = (b << 12) + n;
    float v = pre[c * NPIX + p];
    out[i] = (v - mean[p]) * rstd[p] * g_out[c] + x[i];
}

// ---------------------------------------------------------------------------
extern "C" void kernel_launch(void* const* d_in, const int* in_sizes, int n_in,
                              void* d_out, int out_size, void* d_ws, size_t ws_size,
                              hipStream_t stream) {
    const float* x     = (const float*)d_in[0];   // [16,512,64,64]
    const float* w_qkv = (const float*)d_in[1];   // [1536,512]
    const float* w_out = (const float*)d_in[2];   // [512,512]
    const float* b_out = (const float*)d_in[3];   // [512]
    const float* g_out = (const float*)d_in[4];   // [512]
    const float* g_ln  = (const float*)d_in[5];   // [512]
    float* out = (float*)d_out;

    // workspace carve-up
    char* ws = (char*)d_ws;
    __bf16* qkv_bf  = (__bf16*)ws;                           // 1536 * 65536
    __bf16* attn_bf = qkv_bf + (size_t)CQKV * NPIX;          // 512 * 65536
    __bf16* wq_bf   = attn_bf + (size_t)CIN * NPIX;          // 1536*512
    __bf16* wo_bf   = wq_bf + (size_t)CQKV * CIN;            // 512*512
    uintptr_t a = ((uintptr_t)(wo_bf + (size_t)CIN * CIN) + 255) & ~(uintptr_t)255;
    float* pre   = (float*)a;                                // 512 * 65536
    float* mean1 = pre + (size_t)CIN * NPIX;
    float* rstd1 = mean1 + NPIX;
    float* mean2 = rstd1 + NPIX;
    float* rstd2 = mean2 + NPIX;

    // 1) weight conversion to bf16
    cvt_bf16_kernel<<<(CQKV * CIN + 255) / 256, 256, 0, stream>>>(w_qkv, wq_bf, CQKV * CIN);
    cvt_bf16_kernel<<<(CIN * CIN + 255) / 256, 256, 0, stream>>>(w_out, wo_bf, CIN * CIN);

    // 2) LN stats over channels of x
    ln_stats_x_kernel<<<NPIX / 256, 256, 0, stream>>>(x, mean1, rstd1);

    // 3) fused LN-normalize + QKV GEMM (bf16 WMMA, double-buffered async)
    gemm_qkv_kernel<<<dim3(NPIX / 128, CQKV / 128), 256, 0, stream>>>(
        x, mean1, rstd1, g_ln, wq_bf, qkv_bf);

    // 4) softmaxes (q over d, k over n), in place on bf16 qkv
    softmax_q_kernel<<<(8 * 65536) / 256, 256, 0, stream>>>(qkv_bf);
    softmax_k_kernel<<<16 * 512, 256, 0, stream>>>(qkv_bf);

    // 5) fused context + attention-output GEMMs per (b, h)
    attention_kernel<<<16 * 8, 256, 0, stream>>>(qkv_bf, attn_bf);

    // 6) output projection GEMM + bias (bf16 WMMA, f32 epilogue)
    gemm_out_kernel<<<dim3(NPIX / 128, CIN / 128), 256, 0, stream>>>(
        attn_bf, wo_bf, b_out, pre);

    // 7) final channel LN + residual
    ln_stats_pre_kernel<<<NPIX / 256, 256, 0, stream>>>(pre, mean2, rstd2);
    ln_residual_kernel<<<(16L * 512 * 4096) / 256, 256, 0, stream>>>(
        pre, mean2, rstd2, g_out, x, out);
}
